// ICLAttention_5583457484912
// MI455X (gfx1250) — matrix operand order repaired and checked by reading, not compiled, not run on hardware
//
#include <hip/hip_runtime.h>

// ---------------------------------------------------------------------------
// Problem: out = exclusive_cumsum_s( v @ W_v^T ) with out[:,0,:] = (v@W_v^T)[:,0,:]
//   B=4, S=4096, E=2048  ->  GEMM M=16384, N=2048, K=2048 (137 GFLOP, compute
//   bound at ~480 FLOP/byte), then a bandwidth-bound scan (~400MB).
// GEMM uses bf16x3 split-precision on v_wmma_f32_16x16x32_bf16 (fp32-grade
// accuracy, ~2.6x the FLOP rate of the native 16x16x4 f32 WMMA path).
// ---------------------------------------------------------------------------

typedef __attribute__((ext_vector_type(16))) __bf16 v16bf;
typedef __attribute__((ext_vector_type(8)))  __bf16 v8bf;
typedef __attribute__((ext_vector_type(4)))  __bf16 v4bf;
typedef __attribute__((ext_vector_type(8)))  float  v8f;
typedef __attribute__((ext_vector_type(4)))  float  v4f;

namespace {
constexpr int BATCH = 4;
constexpr int S     = 4096;
constexpr int E     = 2048;
constexpr int M     = BATCH * S;   // 16384
constexpr int N     = E;           // 2048
constexpr int K     = E;           // 2048

constexpr int BM  = 128;
constexpr int BN  = 128;
constexpr int BK  = 32;
constexpr int BKP = 40;            // padded K pitch (bf16 elems) to spread LDS banks

constexpr int SEGLEN = 128;        // scan segment length
constexpr int NSEG   = S / SEGLEN; // 32
} // namespace

#define WMMA_BF16(a, b, c) \
  __builtin_amdgcn_wmma_f32_16x16x32_bf16(false, (a), false, (b), (short)0, (c), false, false)

__device__ __forceinline__ void split4(v4f v, v4bf& h, v4bf& l) {
#pragma unroll
  for (int j = 0; j < 4; ++j) {
    float  x  = v[j];
    __bf16 hb = (__bf16)x;            // RNE truncation to bf16
    h[j] = hb;
    l[j] = (__bf16)(x - (float)hb);   // residual, also bf16
  }
}

__device__ __forceinline__ v16bf cat8(v8bf a, v8bf b) {
  return __builtin_shufflevector(a, b, 0, 1, 2, 3, 4, 5, 6, 7,
                                       8, 9, 10, 11, 12, 13, 14, 15);
}

// ---------------------------------------------------------------------------
// GEMM: C[m,n] = sum_k A[m,k] * W[n,k]   (A = v flattened, W = W_v, C = vp)
// ---------------------------------------------------------------------------
__global__ __launch_bounds__(256, 2)
void gemm_vproj(const float* __restrict__ A, const float* __restrict__ W,
                float* __restrict__ C) {
  __shared__ __bf16 lA_hi[BM][BKP];
  __shared__ __bf16 lA_lo[BM][BKP];
  __shared__ __bf16 lB_hi[BN][BKP];
  __shared__ __bf16 lB_lo[BN][BKP];

  const int t     = threadIdx.x;
  const int lane  = t & 31;
  const int wv    = t >> 5;       // wave id 0..7
  const int waveM = wv >> 1;      // 0..3  (32-row slab)
  const int waveN = wv & 1;       // 0..1  (64-col slab)
  const int half  = lane >> 4;    // lane half selects K sub-range
  const int l16   = lane & 15;

  const int m0 = blockIdx.x * BM;
  const int n0 = blockIdx.y * BN;

  // global staging: thread t loads 4 float4's per matrix per K-step
  const int lr = t >> 3;          // 0..31 base row
  const int lk = (t & 7) * 4;     // 0..28 k offset

  v8f acc[2][4];
#pragma unroll
  for (int mi = 0; mi < 2; ++mi)
#pragma unroll
    for (int ni = 0; ni < 4; ++ni) acc[mi][ni] = (v8f)0.0f;

  v4f ra[4], rb[4];
  auto load_glb = [&](int k0) {
#pragma unroll
    for (int i = 0; i < 4; ++i) {
      ra[i] = *(const v4f*)(A + (size_t)(m0 + lr + 32 * i) * K + k0 + lk);
      rb[i] = *(const v4f*)(W + (size_t)(n0 + lr + 32 * i) * K + k0 + lk);
    }
  };
  auto stage_lds = [&]() {
#pragma unroll
    for (int i = 0; i < 4; ++i) {
      v4bf h, l;
      split4(ra[i], h, l);
      *(v4bf*)&lA_hi[lr + 32 * i][lk] = h;
      *(v4bf*)&lA_lo[lr + 32 * i][lk] = l;
      split4(rb[i], h, l);
      *(v4bf*)&lB_hi[lr + 32 * i][lk] = h;
      *(v4bf*)&lB_lo[lr + 32 * i][lk] = l;
    }
  };

  load_glb(0);

  for (int k0 = 0; k0 < K; k0 += BK) {
    __syncthreads();            // previous compute done: safe to overwrite LDS
    stage_lds();
    __syncthreads();
    if (k0 + BK < K) load_glb(k0 + BK);   // prefetch next slab over the WMMAs

    // --- fragment loads -----------------------------------------------------
    // A (16x32 bf16): lane l16 = M row; VGPR-packed K = {half*8..+7, 16+half*8..+7}
    v16bf a_hi[2], a_lo[2];
#pragma unroll
    for (int mi = 0; mi < 2; ++mi) {
      const int row = waveM * 32 + mi * 16 + l16;
      const __bf16* ph = &lA_hi[row][half * 8];
      const __bf16* pl = &lA_lo[row][half * 8];
      a_hi[mi] = cat8(*(const v8bf*)ph, *(const v8bf*)(ph + 16));
      a_lo[mi] = cat8(*(const v8bf*)pl, *(const v8bf*)(pl + 16));
    }
    // B (32x16 bf16): lane l16 = N col; VGPR-packed K = {half*8..+7, 16+half*8..+7}
    v16bf b_hi[4], b_lo[4];
#pragma unroll
    for (int ni = 0; ni < 4; ++ni) {
      const int col = waveN * 64 + ni * 16 + l16;
      const __bf16* ph = &lB_hi[col][half * 8];
      const __bf16* pl = &lB_lo[col][half * 8];
      b_hi[ni] = cat8(*(const v8bf*)ph, *(const v8bf*)(ph + 16));
      b_lo[ni] = cat8(*(const v8bf*)pl, *(const v8bf*)(pl + 16));
    }

    // --- bf16x3 WMMA: hi*hi + hi*lo + lo*hi (lo*lo ~ 2^-18, dropped) --------
#pragma unroll
    for (int mi = 0; mi < 2; ++mi)
#pragma unroll
      for (int ni = 0; ni < 4; ++ni) {
        acc[mi][ni] = WMMA_BF16(a_hi[mi], b_hi[ni], acc[mi][ni]);
        acc[mi][ni] = WMMA_BF16(a_hi[mi], b_lo[ni], acc[mi][ni]);
        acc[mi][ni] = WMMA_BF16(a_lo[mi], b_hi[ni], acc[mi][ni]);
      }
  }

  // --- epilogue: C/D layout -> lane l16 = N, VGPR r = M row (half*8 + r) -----
#pragma unroll
  for (int mi = 0; mi < 2; ++mi)
#pragma unroll
    for (int ni = 0; ni < 4; ++ni) {
      const int gn = n0 + waveN * 64 + ni * 16 + l16;
#pragma unroll
      for (int r = 0; r < 8; ++r) {
        const int gm = m0 + waveM * 32 + mi * 16 + half * 8 + r;
        C[(size_t)gm * N + gn] = acc[mi][ni][r];
      }
    }
}

// ---------------------------------------------------------------------------
// Scan phase (deterministic, no float atomics): out[b,s,f] = sum_{j<s} vp[b,j,f],
// out[b,0,f] = vp[b,0,f].  In-place on C.
// ---------------------------------------------------------------------------
__global__ __launch_bounds__(256)
void seg_sums(const float* __restrict__ C, float* __restrict__ seg) {
  const int f  = blockIdx.x * 256 + threadIdx.x;
  const int sg = blockIdx.y;
  const int b  = blockIdx.z;
  const float* p = C + ((size_t)b * S + (size_t)sg * SEGLEN) * E + f;
  float s = 0.0f;
#pragma unroll 4
  for (int i = 0; i < SEGLEN; ++i) s += p[(size_t)i * E];
  seg[((size_t)b * NSEG + sg) * E + f] = s;
}

__global__ __launch_bounds__(256)
void seg_scan(float* __restrict__ seg) {
  const int f = blockIdx.x * 256 + threadIdx.x;
  const int b = blockIdx.z;
  float carry = 0.0f;
  for (int sg = 0; sg < NSEG; ++sg) {
    const size_t idx = ((size_t)b * NSEG + sg) * E + f;
    const float t = seg[idx];
    seg[idx] = carry;          // exclusive offsets, in place
    carry += t;
  }
}

__global__ __launch_bounds__(256)
void apply_scan(float* __restrict__ C, const float* __restrict__ seg) {
  const int f  = blockIdx.x * 256 + threadIdx.x;
  const int sg = blockIdx.y;
  const int b  = blockIdx.z;
  float carry = seg[((size_t)b * NSEG + sg) * E + f];
  float* p = C + ((size_t)b * S + (size_t)sg * SEGLEN) * E + f;
  const bool first_seg = (sg == 0);
#pragma unroll 4
  for (int i = 0; i < SEGLEN; ++i) {
    const float v   = p[(size_t)i * E];
    const float out = (first_seg && i == 0) ? v : carry;  // s==0 boundary rule
    carry += v;
    p[(size_t)i * E] = out;
  }
}

// ---------------------------------------------------------------------------
extern "C" void kernel_launch(void* const* d_in, const int* in_sizes, int n_in,
                              void* d_out, int out_size, void* d_ws, size_t ws_size,
                              hipStream_t stream) {
  (void)in_sizes; (void)n_in; (void)out_size; (void)ws_size;
  const float* v  = (const float*)d_in[2];   // [B,S,E]
  const float* Wv = (const float*)d_in[5];   // [E,E]
  float* C   = (float*)d_out;                // [B,S,E]
  float* seg = (float*)d_ws;                 // needs B*NSEG*E*4 = 1 MB

  gemm_vproj<<<dim3(M / BM, N / BN), 256, 0, stream>>>(v, Wv, C);
  seg_sums  <<<dim3(E / 256, NSEG, BATCH), 256, 0, stream>>>(C, seg);
  seg_scan  <<<dim3(E / 256, 1,    BATCH), 256, 0, stream>>>(seg);
  apply_scan<<<dim3(E / 256, NSEG, BATCH), 256, 0, stream>>>(C, seg);
}
